// SparseMoE_19980187861339
// MI455X (gfx1250) — matrix operand order repaired and checked by reading, not compiled, not run on hardware
//
#include <hip/hip_runtime.h>
#include <hip/hip_bf16.h>

#define D_MODEL   1024
#define N_EXPERTS 8
#define HIDDEN    2816
#define NTOK      4096
#define CAPACITY  1024

#define BM 128
#define BN 128
#define BK 32
#define LDSTR 40   // padded LDS row stride in f16 elements (80B: 16B-aligned, bank-spread)

typedef _Float16 half_t;
typedef __attribute__((ext_vector_type(4)))  _Float16 v4h;
typedef __attribute__((ext_vector_type(8)))  _Float16 v8h;
typedef __attribute__((ext_vector_type(16))) _Float16 v16h;
typedef __attribute__((ext_vector_type(8)))  float    v8f;

static __device__ __forceinline__ v16h join8(v8h lo, v8h hi) {
  return __builtin_shufflevector(lo, hi, 0,1,2,3,4,5,6,7,8,9,10,11,12,13,14,15);
}
static __device__ __forceinline__ v4h cvt4(float4 f) {
  return (v4h){(half_t)f.x, (half_t)f.y, (half_t)f.z, (half_t)f.w};
}

// ---------------------------------------------------------------- zero output
__global__ __launch_bounds__(256) void moe_zero_kernel(float4* __restrict__ out, int n4) {
  int i = blockIdx.x * 256 + threadIdx.x;
  if (i < n4) out[i] = make_float4(0.f, 0.f, 0.f, 0.f);
}

// ---------------------------------------------------------------- router: wave per token
__global__ __launch_bounds__(256) void moe_router_kernel(
    const float* __restrict__ x, const float* __restrict__ noise,
    const float* __restrict__ rw, const float* __restrict__ rb,
    const float* __restrict__ nw, const float* __restrict__ nb,
    int* __restrict__ topi, float* __restrict__ gates) {
  int wave = threadIdx.x >> 5;
  int lane = threadIdx.x & 31;
  int tok  = blockIdx.x * 8 + wave;
  float acc[16];
#pragma unroll
  for (int j = 0; j < 16; ++j) acc[j] = 0.f;
  const float* xr = x + (size_t)tok * D_MODEL;
  for (int k = lane; k < D_MODEL; k += 32) {
    float xv = xr[k];
#pragma unroll
    for (int e = 0; e < 8; ++e) {
      acc[e]     += xv * rw[e * D_MODEL + k];
      acc[8 + e] += xv * nw[e * D_MODEL + k];
    }
  }
#pragma unroll
  for (int off = 16; off >= 1; off >>= 1)
#pragma unroll
    for (int j = 0; j < 16; ++j)
      acc[j] += __shfl_xor(acc[j], off, 32);

  if (lane == 0) {
    float noisy[8];
#pragma unroll
    for (int e = 0; e < 8; ++e) {
      float lg = acc[e] + rb[e];
      float nz = acc[8 + e] + nb[e];
      float sp = fmaxf(nz, 0.f) + log1pf(expf(-fabsf(nz)));   // stable softplus
      noisy[e] = lg + noise[tok * 8 + e] * sp;
    }
    int i0 = 0;
#pragma unroll
    for (int e = 1; e < 8; ++e) if (noisy[e] > noisy[i0]) i0 = e;
    int i1 = (i0 == 0) ? 1 : 0;
#pragma unroll
    for (int e = 0; e < 8; ++e) if (e != i0 && noisy[e] > noisy[i1]) i1 = e;
    float g0 = 1.f / (1.f + expf(noisy[i1] - noisy[i0]));     // softmax of top-2
    topi[tok * 2]     = i0;
    topi[tok * 2 + 1] = i1;
    gates[tok * 2]     = g0;
    gates[tok * 2 + 1] = 1.f - g0;
  }
}

// ---------------------------------------------------------------- ordered capacity scan
__global__ __launch_bounds__(1024) void moe_assign_kernel(
    const int* __restrict__ topi, const float* __restrict__ gates,
    int* __restrict__ tokbuf, float* __restrict__ gatebuf) {
  __shared__ int ssum[1024];
  int e = blockIdx.x;
  int t = threadIdx.x;
  int hit[4]; float gt[4]; int cnt = 0;
#pragma unroll
  for (int j = 0; j < 4; ++j) {
    int tok = t * 4 + j;
    int i0 = topi[tok * 2], i1 = topi[tok * 2 + 1];
    bool h0 = (i0 == e), h1 = (i1 == e);
    hit[j] = (h0 || h1) ? 1 : 0;
    gt[j]  = h0 ? gates[tok * 2] : (h1 ? gates[tok * 2 + 1] : 0.f);
    cnt += hit[j];
  }
  ssum[t] = cnt;
  __syncthreads();
  for (int off = 1; off < 1024; off <<= 1) {        // inclusive Hillis-Steele scan
    int v = (t >= off) ? ssum[t - off] : 0;
    __syncthreads();
    ssum[t] += v;
    __syncthreads();
  }
  int pos   = ssum[t] - cnt;                        // exclusive prefix for this thread
  int total = ssum[1023];
#pragma unroll
  for (int j = 0; j < 4; ++j) {
    if (hit[j]) {
      if (pos < CAPACITY) {
        tokbuf[e * CAPACITY + pos]  = t * 4 + j;
        gatebuf[e * CAPACITY + pos] = gt[j];
      }
      pos++;
    }
  }
  if (t >= total) {                                 // empty slots
    tokbuf[e * CAPACITY + t]  = NTOK;
    gatebuf[e * CAPACITY + t] = 0.f;
  }
}

// ---------------------------------------------------------------- gather x -> f16 expert buffers
__global__ __launch_bounds__(256) void moe_gather_kernel(
    const float* __restrict__ x, const int* __restrict__ tokbuf,
    half_t* __restrict__ xb) {
  int s = blockIdx.x, e = blockIdx.y, t = threadIdx.x;
  int tok = tokbuf[e * CAPACITY + s];
  v4h hv;
  if (tok < NTOK) {
    hv = cvt4(*(const float4*)(x + (size_t)tok * D_MODEL + t * 4));
  } else {
    hv = (v4h){(half_t)0.f, (half_t)0.f, (half_t)0.f, (half_t)0.f};
  }
  *(v4h*)(xb + ((size_t)(e * CAPACITY + s)) * D_MODEL + t * 4) = hv;
}

// ---------------------------------------------------------------- GEMM1: h = silu(xb w1^T) * (xb w3^T)
// 512 threads (16 waves, 32x32 each), double-buffered LDS, pipelined staging
__global__ __launch_bounds__(512) void moe_ffn1_kernel(
    const half_t* __restrict__ xb, const float* __restrict__ w1,
    const float* __restrict__ w3, half_t* __restrict__ hbuf) {
  __shared__ half_t As [2][BM * LDSTR];
  __shared__ half_t B1s[2][BN * LDSTR];
  __shared__ half_t B3s[2][BN * LDSTR];
  int e  = blockIdx.z;
  int m0 = blockIdx.y * BM;
  int n0 = blockIdx.x * BN;
  int tid  = threadIdx.x;
  int lane = tid & 31, wave = tid >> 5;
  int wm = wave >> 2, wn = wave & 3;          // 4x4 waves: 32x32 per wave
  int nlane = lane & 15, chalf = lane >> 4;

  const half_t* Ag  = xb + (size_t)e * CAPACITY * D_MODEL;
  const float*  B1g = w1 + (size_t)e * HIDDEN * D_MODEL;
  const float*  B3g = w3 + (size_t)e * HIDDEN * D_MODEL;

  v8f acc1[2][2], acc3[2][2];
#pragma unroll
  for (int mi = 0; mi < 2; ++mi)
#pragma unroll
    for (int ni = 0; ni < 2; ++ni) { acc1[mi][ni] = (v8f)0.f; acc3[mi][ni] = (v8f)0.f; }

  uint4  aR;            // 1 x 16B of A per thread (512 slots total)
  float4 b1R[2], b3R[2];

  auto load_tile = [&](int kk) {
    {
      int r = tid >> 2, q = tid & 3;
      aR = *(const uint4*)(Ag + (size_t)(m0 + r) * D_MODEL + kk + q * 8);
    }
#pragma unroll
    for (int i = 0; i < 2; ++i) {
      int slot = tid + i * 512;
      int r = slot >> 3, q = slot & 7;
      b1R[i] = *(const float4*)(B1g + (size_t)(n0 + r) * D_MODEL + kk + q * 4);
      b3R[i] = *(const float4*)(B3g + (size_t)(n0 + r) * D_MODEL + kk + q * 4);
    }
  };
  auto store_tile = [&](int buf) {
    {
      int r = tid >> 2, q = tid & 3;
      *(uint4*)(&As[buf][r * LDSTR + q * 8]) = aR;
    }
#pragma unroll
    for (int i = 0; i < 2; ++i) {
      int slot = tid + i * 512;
      int r = slot >> 3, q = slot & 7;
      *(v4h*)(&B1s[buf][r * LDSTR + q * 4]) = cvt4(b1R[i]);  // fp32 -> f16 RNE at LDS store
      *(v4h*)(&B3s[buf][r * LDSTR + q * 4]) = cvt4(b3R[i]);
    }
  };

  const int NT = D_MODEL / BK;
  load_tile(0);
  store_tile(0);
  for (int t = 0; t < NT; ++t) {
    __syncthreads();                           // tile t visible; other buffer free
    if (t + 1 < NT) load_tile((t + 1) * BK);   // global loads overlap compute below
    int buf = t & 1;

    v16h bf1[2], bf3[2];
#pragma unroll
    for (int ni = 0; ni < 2; ++ni) {       // B frag: lane n<16 -> col n, K0..15; lane n+16 -> K16..31
      int n = wn * 32 + ni * 16 + nlane;
      bf1[ni] = *(const v16h*)(&B1s[buf][n * LDSTR + chalf * 16]);
      bf3[ni] = *(const v16h*)(&B3s[buf][n * LDSTR + chalf * 16]);
    }
#pragma unroll
    for (int mi = 0; mi < 2; ++mi) {       // A frag: lane m<16 -> K0..15; lane m+16 -> K16..31
      int m = wm * 32 + mi * 16 + nlane;
      v8h lo = *(const v8h*)(&As[buf][m * LDSTR + chalf * 16]);
      v8h hi = *(const v8h*)(&As[buf][m * LDSTR + chalf * 16 + 8]);
      v16h af = join8(lo, hi);
#pragma unroll
      for (int ni = 0; ni < 2; ++ni) {
        acc1[mi][ni] = __builtin_amdgcn_wmma_f32_16x16x32_f16(false, af, false, bf1[ni], (short)0, acc1[mi][ni], false, false);
        acc3[mi][ni] = __builtin_amdgcn_wmma_f32_16x16x32_f16(false, af, false, bf3[ni], (short)0, acc3[mi][ni], false, false);
      }
    }
    if (t + 1 < NT) store_tile((t + 1) & 1);
  }
  // epilogue: h = silu(c1) * c3 -> f16
#pragma unroll
  for (int mi = 0; mi < 2; ++mi)
#pragma unroll
    for (int ni = 0; ni < 2; ++ni)
#pragma unroll
      for (int i = 0; i < 8; ++i) {
        int m = m0 + wm * 32 + mi * 16 + chalf * 8 + i;   // C: vgpr i -> M=i (+8 for hi lanes)
        int n = n0 + wn * 32 + ni * 16 + nlane;
        float c1 = acc1[mi][ni][i], c3 = acc3[mi][ni][i];
        float hv = (c1 / (1.f + expf(-c1))) * c3;
        hbuf[((size_t)e * CAPACITY + m) * HIDDEN + n] = (half_t)hv;
      }
}

// ---------------------------------------------------------------- GEMM2: out += gate * (h w2^T)
__global__ __launch_bounds__(512) void moe_ffn2_kernel(
    const half_t* __restrict__ hbuf, const float* __restrict__ w2,
    const int* __restrict__ tokbuf, const float* __restrict__ gatebuf,
    float* __restrict__ out) {
  __shared__ half_t As[2][BM * LDSTR];
  __shared__ half_t Bs[2][BN * LDSTR];
  int e  = blockIdx.z;
  int m0 = blockIdx.y * BM;
  int n0 = blockIdx.x * BN;
  int tid  = threadIdx.x;
  int lane = tid & 31, wave = tid >> 5;
  int wm = wave >> 2, wn = wave & 3;
  int nlane = lane & 15, chalf = lane >> 4;

  const half_t* Ag = hbuf + (size_t)e * CAPACITY * HIDDEN;
  const float*  Bg = w2   + (size_t)e * D_MODEL * HIDDEN;

  v8f acc[2][2];
#pragma unroll
  for (int mi = 0; mi < 2; ++mi)
#pragma unroll
    for (int ni = 0; ni < 2; ++ni) acc[mi][ni] = (v8f)0.f;

  uint4  aR;
  float4 bR[2];

  auto load_tile = [&](int kk) {
    {
      int r = tid >> 2, q = tid & 3;
      aR = *(const uint4*)(Ag + (size_t)(m0 + r) * HIDDEN + kk + q * 8);
    }
#pragma unroll
    for (int i = 0; i < 2; ++i) {
      int slot = tid + i * 512;
      int r = slot >> 3, q = slot & 7;
      bR[i] = *(const float4*)(Bg + (size_t)(n0 + r) * HIDDEN + kk + q * 4);
    }
  };
  auto store_tile = [&](int buf) {
    {
      int r = tid >> 2, q = tid & 3;
      *(uint4*)(&As[buf][r * LDSTR + q * 8]) = aR;
    }
#pragma unroll
    for (int i = 0; i < 2; ++i) {
      int slot = tid + i * 512;
      int r = slot >> 3, q = slot & 7;
      *(v4h*)(&Bs[buf][r * LDSTR + q * 4]) = cvt4(bR[i]);
    }
  };

  const int NT = HIDDEN / BK;
  load_tile(0);
  store_tile(0);
  for (int t = 0; t < NT; ++t) {
    __syncthreads();
    if (t + 1 < NT) load_tile((t + 1) * BK);
    int buf = t & 1;

    v16h bf[2];
#pragma unroll
    for (int ni = 0; ni < 2; ++ni) {
      int n = wn * 32 + ni * 16 + nlane;
      bf[ni] = *(const v16h*)(&Bs[buf][n * LDSTR + chalf * 16]);
    }
#pragma unroll
    for (int mi = 0; mi < 2; ++mi) {
      int m = wm * 32 + mi * 16 + nlane;
      v8h lo = *(const v8h*)(&As[buf][m * LDSTR + chalf * 16]);
      v8h hi = *(const v8h*)(&As[buf][m * LDSTR + chalf * 16 + 8]);
      v16h af = join8(lo, hi);
#pragma unroll
      for (int ni = 0; ni < 2; ++ni)
        acc[mi][ni] = __builtin_amdgcn_wmma_f32_16x16x32_f16(false, af, false, bf[ni], (short)0, acc[mi][ni], false, false);
    }
    if (t + 1 < NT) store_tile((t + 1) & 1);
  }
  // epilogue: gate-scaled scatter-add (token appears in <=2 experts)
#pragma unroll
  for (int mi = 0; mi < 2; ++mi) {
#pragma unroll
    for (int i = 0; i < 8; ++i) {
      int mrow = m0 + wm * 32 + mi * 16 + chalf * 8 + i;
      int tok  = tokbuf[e * CAPACITY + mrow];
      if (tok < NTOK) {
        float g = gatebuf[e * CAPACITY + mrow];
#pragma unroll
        for (int ni = 0; ni < 2; ++ni) {
          int n = n0 + wn * 32 + ni * 16 + nlane;
          atomicAdd(out + (size_t)tok * D_MODEL + n, acc[mi][ni][i] * g);
        }
      }
    }
  }
}

// ---------------------------------------------------------------- launch
extern "C" void kernel_launch(void* const* d_in, const int* in_sizes, int n_in,
                              void* d_out, int out_size, void* d_ws, size_t ws_size,
                              hipStream_t stream) {
  const float* x     = (const float*)d_in[0];
  const float* noise = (const float*)d_in[1];
  const float* rw    = (const float*)d_in[2];
  const float* rb    = (const float*)d_in[3];
  const float* nw    = (const float*)d_in[4];
  const float* nb    = (const float*)d_in[5];
  const float* w1    = (const float*)d_in[6];
  const float* w3    = (const float*)d_in[7];
  const float* w2    = (const float*)d_in[8];
  float* out = (float*)d_out;

  // workspace layout (~60.2 MB total)
  char* ws = (char*)d_ws;
  int*    topi    = (int*)  (ws);                    // 32 KB
  float*  gates   = (float*)(ws + (32 << 10));       // 32 KB
  int*    tokbuf  = (int*)  (ws + (64 << 10));       // 32 KB
  float*  gatebuf = (float*)(ws + (96 << 10));       // 32 KB
  half_t* xb      = (half_t*)(ws + (128 << 10));     // 16 MB
  half_t* hbuf    = (half_t*)(ws + (128 << 10) + (size_t)N_EXPERTS * CAPACITY * D_MODEL * 2); // 44 MB

  moe_zero_kernel<<<(NTOK * D_MODEL / 4 + 255) / 256, 256, 0, stream>>>((float4*)out, NTOK * D_MODEL / 4);
  moe_router_kernel<<<NTOK / 8, 256, 0, stream>>>(x, noise, rw, rb, nw, nb, topi, gates);
  moe_assign_kernel<<<N_EXPERTS, 1024, 0, stream>>>(topi, gates, tokbuf, gatebuf);
  moe_gather_kernel<<<dim3(CAPACITY, N_EXPERTS), 256, 0, stream>>>(x, tokbuf, xb);
  moe_ffn1_kernel<<<dim3(HIDDEN / BN, CAPACITY / BM, N_EXPERTS), 512, 0, stream>>>(xb, w1, w3, hbuf);
  moe_ffn2_kernel<<<dim3(D_MODEL / BN, CAPACITY / BM, N_EXPERTS), 512, 0, stream>>>(hbuf, w2, tokbuf, gatebuf, out);
}
